// Rwkv7SelfAttention_39341900432039
// MI455X (gfx1250) — compile-verified, matrix operand order checked by reading, not run on hardware
//
#include <hip/hip_runtime.h>
#include <math.h>

// ---------------------------------------------------------------------------
// RWKV7 self-attention block for gfx1250 (MI455X), wave32 / WMMA.
//   - GEMMs: v_wmma_f32_16x16x32_bf16; MIX/epilogue/tile are template params.
//     Double-buffered LDS (one barrier per K-step, staging overlaps WMMA).
//     128x64 block tile -> 4 WMMA per wave per K-step for the 2048-wide GEMMs.
//   - Recurrent scan: rank-1 ab => two matvecs/step, state rows in VGPRs.
// Workspace: ~100 MB fp32 in d_ws.
// ---------------------------------------------------------------------------

constexpr int HIDDEN = 2048;
constexpr int TSEQ   = 1024;
constexpr int HEAD   = 64;
constexpr int NH     = HIDDEN / HEAD;   // 32

typedef __attribute__((ext_vector_type(16))) __bf16 v16bf;
typedef __attribute__((ext_vector_type(8)))  __bf16 v8bf;
typedef __attribute__((ext_vector_type(4)))  __bf16 v4bf;
typedef __attribute__((ext_vector_type(8)))  float  v8f;

__device__ __forceinline__ __bf16 f2bf(float f) {
  unsigned int u = __builtin_bit_cast(unsigned int, f);
  u += 0x7FFFu + ((u >> 16) & 1u);          // round-to-nearest-even
  unsigned short h = (unsigned short)(u >> 16);
  return __builtin_bit_cast(__bf16, h);
}

__device__ __forceinline__ float sigm(float x) { return 1.0f / (1.0f + __expf(-x)); }

__device__ __forceinline__ float block_reduce64(float v, float* red, int tid) {
  red[tid] = v;
  __syncthreads();
#pragma unroll
  for (int o = 32; o > 0; o >>= 1) {
    if (tid < o) red[tid] += red[tid + o];
    __syncthreads();
  }
  float r = red[0];
  __syncthreads();
  return r;
}

// ---------------------------------------------------------------------------
// LayerNorm over HIDDEN per token; also emits state1_out = xn[T-1].
// ---------------------------------------------------------------------------
__global__ __launch_bounds__(256)
void ln_prep(const float* __restrict__ x, const float* __restrict__ g,
             const float* __restrict__ b, float* __restrict__ xn,
             float* __restrict__ state1_out) {
  __shared__ float red[256];
  const int t = blockIdx.x, tid = threadIdx.x;
  const float* xr = x + (size_t)t * HIDDEN;
  float vals[8];
  float s = 0.f, sq = 0.f;
#pragma unroll
  for (int i = 0; i < 8; ++i) {
    float v = xr[tid + i * 256];
    vals[i] = v; s += v; sq += v * v;
  }
  red[tid] = s; __syncthreads();
  for (int o = 128; o > 0; o >>= 1) { if (tid < o) red[tid] += red[tid + o]; __syncthreads(); }
  const float mean = red[0] * (1.0f / HIDDEN);
  __syncthreads();
  red[tid] = sq; __syncthreads();
  for (int o = 128; o > 0; o >>= 1) { if (tid < o) red[tid] += red[tid + o]; __syncthreads(); }
  const float var  = red[0] * (1.0f / HIDDEN) - mean * mean;
  const float rstd = rsqrtf(var + 1e-5f);
  float* o = xn + (size_t)t * HIDDEN;
#pragma unroll
  for (int i = 0; i < 8; ++i) {
    int c = tid + i * 256;
    float v = (vals[i] - mean) * rstd * g[c] + b[c];
    o[c] = v;
    if (t == TSEQ - 1) state1_out[c] = v;
  }
}

// sx[t] = (t==0 ? state1 : xn[t-1]) - xn[t]
__global__ __launch_bounds__(256)
void sx_kernel(const float* __restrict__ xn, const float* __restrict__ state1,
               float* __restrict__ sx) {
  const int t = blockIdx.x, tid = threadIdx.x;
#pragma unroll
  for (int i = 0; i < 8; ++i) {
    int c = tid + i * 256;
    float prev = (t == 0) ? state1[c] : xn[(size_t)(t - 1) * HIDDEN + c];
    sx[(size_t)t * HIDDEN + c] = prev - xn[(size_t)t * HIDDEN + c];
  }
}

// ---------------------------------------------------------------------------
// Tiled GEMM: C[M,N] = epilogue( mix(A,Asx,coef)[M,K] @ W[K,N] )
// Block = 256 thr (8 waves), tile TMT(M) x TNT(N), K-step 32, double-buffered
// LDS (one barrier per K-step; next tile's global loads overlap the WMMAs).
// TMT=128,TNT=64: waves 4x2, each wave 32x32 = 4 WMMA (2 A-frag x 2 B-frag).
// TMT= 64,TNT=32: waves 4x2, each wave 16x16 = 1 WMMA.
// EP: 0 none | 1 tanh | 2 sigmoid(bias+x) | 3 exp(-.6065*sigmoid(bias+x))
//     4 residual + x     (all compile-time; no branches in hot code)
// ---------------------------------------------------------------------------
constexpr int TK = 32, LDA = 40;

template <bool MIX, int EP, int TMT, int TNT>
__global__ __launch_bounds__(256)
void gemm_bf16_wmma(const float* __restrict__ A, const float* __restrict__ Asx,
                    const float* __restrict__ coef, const float* __restrict__ W,
                    const float* __restrict__ bias, const float* __restrict__ residual,
                    float* __restrict__ C, int N, int K) {
  constexpr int MF = TMT / 64;             // A fragments per wave (2 or 1)
  constexpr int NF = TNT / 32;             // B fragments per wave (2 or 1)
  __shared__ __bf16 As[2][TMT][LDA];
  __shared__ __bf16 Bs[2][TNT][LDA];       // transposed: Bs[buf][n][k]

  const int tid = threadIdx.x;
  const int m0  = blockIdx.y * TMT;
  const int n0  = blockIdx.x * TNT;
  const int lane = tid & 31;
  const int wave = tid >> 5;
  const int wm = wave >> 1;                // 0..3
  const int wn = wave & 1;                 // 0..1
  const int lr = lane & 15;
  const int kh = (lane >> 4) << 3;         // 0 or 8 (ISA 16b fragment layout)

  // stage one K-tile (A: TMTx32 with fused token-mix, B: 32xTNT K-transposed)
  auto stage = [&](int k0, int buf) {
#pragma unroll
    for (int i = 0; i < (TMT * TK) / (256 * 4); ++i) {
      int ci = tid + i * 256;
      int row = ci >> 3;
      int c4  = (ci & 7) << 2;
      size_t gi = (size_t)(m0 + row) * K + (k0 + c4);
      float4 va = *(const float4*)(A + gi);
      if constexpr (MIX) {
        float4 sv = *(const float4*)(Asx + gi);
        float4 cf = *(const float4*)(coef + k0 + c4);
        va.x += cf.x * sv.x; va.y += cf.y * sv.y;
        va.z += cf.z * sv.z; va.w += cf.w * sv.w;
      }
      v4bf pb; pb[0] = f2bf(va.x); pb[1] = f2bf(va.y);
               pb[2] = f2bf(va.z); pb[3] = f2bf(va.w);
      *(v4bf*)&As[buf][row][c4] = pb;
    }
    constexpr int BCH = TNT / 4;           // float4 chunks per k-row
#pragma unroll
    for (int i = 0; i < (TK * TNT) / (256 * 4); ++i) {
      int ci = tid + i * 256;
      int kr  = ci / BCH;
      int nc4 = (ci % BCH) << 2;
      size_t gi = (size_t)(k0 + kr) * N + (n0 + nc4);
      float4 vb = *(const float4*)(W + gi);
      Bs[buf][nc4 + 0][kr] = f2bf(vb.x);
      Bs[buf][nc4 + 1][kr] = f2bf(vb.y);
      Bs[buf][nc4 + 2][kr] = f2bf(vb.z);
      Bs[buf][nc4 + 3][kr] = f2bf(vb.w);
    }
  };

  v8f acc[MF][NF];
#pragma unroll
  for (int fm = 0; fm < MF; ++fm)
#pragma unroll
    for (int fn = 0; fn < NF; ++fn) acc[fm][fn] = v8f{};

  stage(0, 0);
  int p = 0;
  for (int k0 = 0; k0 < K; k0 += TK) {
    __syncthreads();

    // fragments from buffer p: contiguous 8xbf16 runs -> ds_load_b128
    v16bf af[MF], bfr[NF];
#pragma unroll
    for (int fm = 0; fm < MF; ++fm) {
      const int rr = wm * 16 * MF + fm * 16 + lr;
      const v8bf lo = *(const v8bf*)&As[p][rr][kh];
      const v8bf hi = *(const v8bf*)&As[p][rr][kh + 16];
#pragma unroll
      for (int e = 0; e < 8; ++e) { af[fm][e] = lo[e]; af[fm][e + 8] = hi[e]; }
    }
#pragma unroll
    for (int fn = 0; fn < NF; ++fn) {
      const int nr = wn * 16 * NF + fn * 16 + lr;
      const v8bf lo = *(const v8bf*)&Bs[p][nr][kh];
      const v8bf hi = *(const v8bf*)&Bs[p][nr][kh + 16];
#pragma unroll
      for (int e = 0; e < 8; ++e) { bfr[fn][e] = lo[e]; bfr[fn][e + 8] = hi[e]; }
    }

    // stage next tile into the other buffer (overlaps the WMMAs below)
    if (k0 + TK < K) stage(k0 + TK, p ^ 1);
    if (k0 + 2 * TK < K)                   // prefetch tile after next
      __builtin_prefetch(W + (size_t)(k0 + 2 * TK + (tid >> 3)) * N + n0 + ((tid & 7) << 2), 0, 1);

#pragma unroll
    for (int fm = 0; fm < MF; ++fm)
#pragma unroll
      for (int fn = 0; fn < NF; ++fn)
        acc[fm][fn] = __builtin_amdgcn_wmma_f32_16x16x32_bf16(
            false, af[fm], false, bfr[fn], (short)0, acc[fm][fn], false, false);

    p ^= 1;
  }

  // --- epilogue; D layout: col = lane&15, rows = 8*(lane>>4)+e ---
#pragma unroll
  for (int fm = 0; fm < MF; ++fm) {
    const int rbase = m0 + wm * 16 * MF + fm * 16 + ((lane >> 4) << 3);
#pragma unroll
    for (int fn = 0; fn < NF; ++fn) {
      const int col = n0 + wn * 16 * NF + fn * 16 + lr;
      float bv = 0.0f;
      if constexpr (EP == 2 || EP == 3) { if (bias) bv = bias[col]; }
#pragma unroll
      for (int e = 0; e < 8; ++e) {
        size_t idx = (size_t)(rbase + e) * N + col;
        float v = acc[fm][fn][e];
        if constexpr (EP == 1) v = tanhf(v);
        else if constexpr (EP == 2) v = sigm(v + bv);
        else if constexpr (EP == 3) v = __expf(-0.606531f * sigm(bv + v));
        else if constexpr (EP == 4) v = residual[idx] + v;
        C[idx] = v;
      }
    }
  }
}

// ---------------------------------------------------------------------------
// Per-(t,head): kk = normalize(k*k_k); k *= 1+(a-1)*k_a; v = lerp(v, v_first, vsig)
// ---------------------------------------------------------------------------
__global__ __launch_bounds__(64)
void postk(float* __restrict__ k, float* __restrict__ v,
           const float* __restrict__ a, const float* __restrict__ vsig,
           const float* __restrict__ v_first, const float* __restrict__ k_k,
           const float* __restrict__ k_a, float* __restrict__ kk) {
  __shared__ float red[64];
  const int h = blockIdx.x, t = blockIdx.y, i = threadIdx.x;
  const int c = h * HEAD + i;
  const size_t idx = (size_t)t * HIDDEN + c;
  const float kv = k[idx];
  const float kkraw = kv * k_k[c];
  const float ss = block_reduce64(kkraw * kkraw, red, i);
  kk[idx] = kkraw / fmaxf(sqrtf(ss), 1e-12f);
  k[idx] = kv * (1.0f + (a[idx] - 1.0f) * k_a[c]);
  const float vv = v[idx];
  v[idx] = vv + (v_first[idx] - vv) * vsig[idx];
}

// ---------------------------------------------------------------------------
// Recurrent scan. One block (2 waves) per head; thread i owns state row i
// (64 f32 in VGPRs). Rank-1 ab => sa = S.kk, then fused decay/update/readout.
// ---------------------------------------------------------------------------
__global__ __launch_bounds__(64)
void scan_kernel(const float* __restrict__ w, const float* __restrict__ kk,
                 const float* __restrict__ a, const float* __restrict__ k,
                 const float* __restrict__ v, const float* __restrict__ r,
                 const float* __restrict__ state2, float* __restrict__ y,
                 float* __restrict__ S_out) {
  __shared__ float sw[HEAD], skk[HEAD], skka[HEAD], sk[HEAD], sv[HEAD], sr[HEAD];
  const int h = blockIdx.x, i = threadIdx.x;
  float S[HEAD];
#pragma unroll
  for (int j = 0; j < HEAD; ++j)
    S[j] = state2[(size_t)h * HEAD * HEAD + (size_t)i * HEAD + j];

  for (int t = 0; t < TSEQ; ++t) {
    const size_t base = (size_t)t * HIDDEN + (size_t)h * HEAD;
    const float kkv = kk[base + i];
    sw[i] = w[base + i];
    skk[i] = kkv;
    skka[i] = kkv * a[base + i];
    sk[i] = k[base + i];
    sv[i] = v[base + i];
    sr[i] = r[base + i];
    __syncthreads();

    float sa = 0.f;
#pragma unroll 8
    for (int j = 0; j < HEAD; ++j) sa += S[j] * skk[j];

    const float vi = sv[i];
    float yi = 0.f;
#pragma unroll 8
    for (int j = 0; j < HEAD; ++j) {
      float s = S[j] * sw[j] - sa * skka[j] + vi * sk[j];
      S[j] = s;
      yi += s * sr[j];
    }
    y[base + i] = yi;
    __syncthreads();
  }
#pragma unroll
  for (int j = 0; j < HEAD; ++j)
    S_out[(size_t)h * HEAD * HEAD + (size_t)i * HEAD + j] = S[j];
}

// ---------------------------------------------------------------------------
// Per-(t,head): group-norm(y) + rkv, gate by g -> z
// ---------------------------------------------------------------------------
__global__ __launch_bounds__(64)
void mix2(const float* __restrict__ y, const float* __restrict__ r,
          const float* __restrict__ k, const float* __restrict__ v,
          const float* __restrict__ g, const float* __restrict__ r_k,
          const float* __restrict__ lnw, const float* __restrict__ lnb,
          float* __restrict__ z) {
  __shared__ float red[64];
  const int h = blockIdx.x, t = blockIdx.y, i = threadIdx.x;
  const int c = h * HEAD + i;
  const size_t idx = (size_t)t * HIDDEN + c;
  const float yv = y[idx];
  const float mean = block_reduce64(yv, red, i) * (1.0f / HEAD);
  const float d = yv - mean;
  const float var = block_reduce64(d * d, red, i) * (1.0f / HEAD);
  const float yln = d * rsqrtf(var + 0.00064f) * lnw[c] + lnb[c];
  const float rkvs = block_reduce64(r[idx] * k[idx] * r_k[c], red, i);
  z[idx] = (yln + rkvs * v[idx]) * g[idx];
}

// ---------------------------------------------------------------------------
extern "C" void kernel_launch(void* const* d_in, const int* in_sizes, int n_in,
                              void* d_out, int out_size, void* d_ws, size_t ws_size,
                              hipStream_t stream) {
  (void)in_sizes; (void)n_in; (void)out_size; (void)ws_size;
  const float* x       = (const float*)d_in[0];
  const float* state1  = (const float*)d_in[1];
  const float* state2  = (const float*)d_in[2];
  const float* v_first = (const float*)d_in[3];
  const float* ln1_w   = (const float*)d_in[4];
  const float* ln1_b   = (const float*)d_in[5];
  const float* x_r     = (const float*)d_in[6];
  const float* x_w     = (const float*)d_in[7];
  const float* x_k     = (const float*)d_in[8];
  const float* x_v     = (const float*)d_in[9];
  const float* x_a     = (const float*)d_in[10];
  const float* x_g     = (const float*)d_in[11];
  const float* Wr      = (const float*)d_in[12];
  const float* Wk      = (const float*)d_in[13];
  const float* Wv      = (const float*)d_in[14];
  const float* Wo      = (const float*)d_in[15];
  const float* w0      = (const float*)d_in[16];
  const float* w1      = (const float*)d_in[17];
  const float* w2      = (const float*)d_in[18];
  const float* a0      = (const float*)d_in[19];
  const float* a1      = (const float*)d_in[20];
  const float* a2      = (const float*)d_in[21];
  const float* v0      = (const float*)d_in[22];
  const float* v1      = (const float*)d_in[23];
  const float* v2      = (const float*)d_in[24];
  const float* g1      = (const float*)d_in[25];
  const float* g2      = (const float*)d_in[26];
  const float* k_k     = (const float*)d_in[27];
  const float* k_a     = (const float*)d_in[28];
  const float* r_k     = (const float*)d_in[29];
  const float* ln_x_w  = (const float*)d_in[30];
  const float* ln_x_b  = (const float*)d_in[31];

  const size_t TH = (size_t)TSEQ * HIDDEN;
  float* ws   = (float*)d_ws;
  float* xn   = ws + 0 * TH;
  float* sx   = ws + 1 * TH;
  float* rb   = ws + 2 * TH;
  float* kb   = ws + 3 * TH;
  float* vb   = ws + 4 * TH;
  float* wb   = ws + 5 * TH;
  float* ab   = ws + 6 * TH;
  float* gb   = ws + 7 * TH;
  float* vsg  = ws + 8 * TH;
  float* kkb  = ws + 9 * TH;
  float* yb   = ws + 10 * TH;
  float* zb   = ws + 11 * TH;
  float* wmid = ws + 12 * TH;
  float* amid = wmid + (size_t)TSEQ * 96;
  float* vmid = amid + (size_t)TSEQ * 96;
  float* gmid = vmid + (size_t)TSEQ * 64;

  float* outp = (float*)d_out;            // (out, state1, S_final, v_first)
  float* s1o  = outp + TH;
  float* Sfo  = s1o + HIDDEN;
  float* vfo  = Sfo + (size_t)NH * HEAD * HEAD;

  ln_prep<<<TSEQ, 256, 0, stream>>>(x, ln1_w, ln1_b, xn, s1o);
  sx_kernel<<<TSEQ, 256, 0, stream>>>(xn, state1, sx);

  const dim3 gBig(HIDDEN / 64, TSEQ / 128);     // 128x64 tiles
  // r/k/v projections (token-mix fused)
  gemm_bf16_wmma<true, 0, 128, 64><<<gBig, 256, 0, stream>>>(xn, sx, x_r, Wr, nullptr, nullptr, rb, HIDDEN, HIDDEN);
  gemm_bf16_wmma<true, 0, 128, 64><<<gBig, 256, 0, stream>>>(xn, sx, x_k, Wk, nullptr, nullptr, kb, HIDDEN, HIDDEN);
  gemm_bf16_wmma<true, 0, 128, 64><<<gBig, 256, 0, stream>>>(xn, sx, x_v, Wv, nullptr, nullptr, vb, HIDDEN, HIDDEN);
  // LoRA-down projections (activation fused); N=96/64 -> 64x32 tiles
  gemm_bf16_wmma<true, 1, 64, 32><<<dim3(96 / 32, TSEQ / 64), 256, 0, stream>>>(xn, sx, x_w, w1, nullptr, nullptr, wmid, 96, HIDDEN);
  gemm_bf16_wmma<true, 0, 64, 32><<<dim3(96 / 32, TSEQ / 64), 256, 0, stream>>>(xn, sx, x_a, a1, nullptr, nullptr, amid, 96, HIDDEN);
  gemm_bf16_wmma<true, 0, 64, 32><<<dim3(64 / 32, TSEQ / 64), 256, 0, stream>>>(xn, sx, x_v, v1, nullptr, nullptr, vmid, 64, HIDDEN);
  gemm_bf16_wmma<true, 2, 128, 64><<<dim3(256 / 64, TSEQ / 128), 256, 0, stream>>>(xn, sx, x_g, g1, nullptr, nullptr, gmid, 256, HIDDEN);
  // LoRA-up projections (gate nonlinearity fused)
  gemm_bf16_wmma<false, 3, 128, 64><<<gBig, 256, 0, stream>>>(wmid, nullptr, nullptr, w2, w0, nullptr, wb, HIDDEN, 96);
  gemm_bf16_wmma<false, 2, 128, 64><<<gBig, 256, 0, stream>>>(amid, nullptr, nullptr, a2, a0, nullptr, ab, HIDDEN, 96);
  gemm_bf16_wmma<false, 2, 128, 64><<<gBig, 256, 0, stream>>>(vmid, nullptr, nullptr, v2, v0, nullptr, vsg, HIDDEN, 64);
  gemm_bf16_wmma<false, 0, 128, 64><<<gBig, 256, 0, stream>>>(gmid, nullptr, nullptr, g2, nullptr, nullptr, gb, HIDDEN, 256);

  postk<<<dim3(NH, TSEQ), 64, 0, stream>>>(kb, vb, ab, vsg, v_first, k_k, k_a, kkb);
  scan_kernel<<<NH, 64, 0, stream>>>(wb, kkb, ab, kb, vb, rb, state2, yb, Sfo);
  mix2<<<dim3(NH, TSEQ), 64, 0, stream>>>(yb, rb, kb, vb, gb, r_k, ln_x_w, ln_x_b, zb);

  // out = x + z @ Wo  (residual fused)
  gemm_bf16_wmma<false, 4, 128, 64><<<gBig, 256, 0, stream>>>(zb, nullptr, nullptr, Wo, nullptr, x, outp, HIDDEN, HIDDEN);

  // v_first passthrough output
  hipMemcpyAsync(vfo, v_first, TH * sizeof(float), hipMemcpyDeviceToDevice, stream);
}
